// PointTransformerLayer_32134945308730
// MI455X (gfx1250) — matrix operand order repaired and checked by reading, not compiled, not run on hardware
//
#include <hip/hip_runtime.h>

typedef __attribute__((ext_vector_type(16))) _Float16 v16h;
typedef __attribute__((ext_vector_type(8)))  _Float16 v8h;
typedef __attribute__((ext_vector_type(8)))  float    v8f;

#define STRH 72      // f16 row stride in halves (144B: 16B-aligned, bank-safe)
#define STRG 66      // f32 g/k-matrix row stride in floats
#define JCH  128     // j-chunk processed per LDS pass
#define NTHREADS 256

// ---- WMMA fragment loaders (CDNA5 ISA 7.12.2 layouts) ----
// A (16x32 f16): lanes 0-15 hold row M=lane, halves K=kb..kb+7 and kb+16..kb+23;
//                lanes 16-31 hold row M=lane-16, halves K=kb+8..15 and kb+24..31.
__device__ __forceinline__ v16h load_fragA(const _Float16* __restrict__ buf,
                                           int m0, int kbase, int lane) {
  int r  = m0 + (lane & 15);
  int kq = kbase + ((lane >> 4) << 3);
  const _Float16* p = buf + r * STRH + kq;
  v8h lo = *(const v8h*)(p);
  v8h hi = *(const v8h*)(p + 16);
  v16h a;
#pragma unroll
  for (int t = 0; t < 8; ++t) { a[t] = lo[t]; a[t + 8] = hi[t]; }
  return a;
}
// B (32x16 f16): lanes 0-15 hold K=kb..kb+7 and kb+16..kb+23 at col N=lane; lanes 16-31 hold
// K=kb+8..kb+15 and kb+24..kb+31 at col N=lane-16. Weights stored transposed (N-major) in LDS.
__device__ __forceinline__ v16h load_fragB(const _Float16* __restrict__ wt,
                                           int n0, int kbase, int lane) {
  int n = n0 + (lane & 15);
  int k = kbase + ((lane >> 4) << 3);
  const _Float16* p = wt + n * STRH + k;
  v8h lo = *(const v8h*)(p);
  v8h hi = *(const v8h*)(p + 16);
  v16h b;
#pragma unroll
  for (int t = 0; t < 8; ++t) { b[t] = lo[t]; b[t + 8] = hi[t]; }
  return b;
}

// ---- Kernel 1: BatchNorm stats over pairwise deltas, closed form ----
// mean_c over (x_i - x_j) is exactly 0; var_c = (1/BN^2) * sum_b (2N*Q_bc - 2*S_bc^2)
__global__ void bn_stats_kernel(const float* __restrict__ xyz,
                                const float* __restrict__ gamma,
                                const float* __restrict__ beta,
                                float* __restrict__ scws) {
  __shared__ float part[12];
  int t = threadIdx.x;
  if (t < 12) {
    int b = t / 3, c = t % 3;
    float S = 0.f, Q = 0.f;
    for (int i = 0; i < 512; ++i) {
      float x = xyz[((size_t)b * 512 + i) * 3 + c];
      S += x; Q = fmaf(x, x, Q);
    }
    part[t] = 2.f * 512.f * Q - 2.f * S * S;
  }
  __syncthreads();
  if (t < 3) {
    float var = 0.f;
    for (int b = 0; b < 4; ++b) var += part[b * 3 + t];
    var *= (1.f / (4.f * 512.f * 512.f));
    scws[t]     = gamma[t] * rsqrtf(var + 1e-5f);  // scale
    scws[4 + t] = beta[t];                          // shift (mean == 0)
  }
}

// ---- Kernel 2: q/k/v projections [2048,64] x [64,64] x3 ----
__global__ void qkv_kernel(const float* __restrict__ points,
                           const float* __restrict__ Wq, const float* __restrict__ bq,
                           const float* __restrict__ Wk, const float* __restrict__ bk,
                           const float* __restrict__ Wv, const float* __restrict__ bv,
                           float* __restrict__ q, float* __restrict__ k,
                           float* __restrict__ v) {
  __shared__ float row[64];
  int r = blockIdx.x;          // point row 0..2047
  int t = threadIdx.x;         // 0..191
  if (t < 64) row[t] = points[(size_t)r * 64 + t];
  __syncthreads();
  int which = t >> 6, m = t & 63;
  const float* W  = (which == 0) ? Wq : (which == 1) ? Wk : Wv;
  const float* bb = (which == 0) ? bq : (which == 1) ? bk : bv;
  float acc = bb[m];
#pragma unroll 8
  for (int c = 0; c < 64; ++c) acc = fmaf(row[c], W[c * 64 + m], acc);
  float* dst = (which == 0) ? q : (which == 1) ? k : v;
  dst[(size_t)r * 64 + m] = acc;
}

// ---- Kernel 3: fused point-transformer row, one WG per (b,i) ----
__global__ __launch_bounds__(NTHREADS)
void pt_layer_kernel(const float* __restrict__ xyz,
                     const float* __restrict__ W_p1, const float* __restrict__ b_p1,
                     const float* __restrict__ W_p2, const float* __restrict__ b_p2,
                     const float* __restrict__ Wg1,  const float* __restrict__ bg1,
                     const float* __restrict__ Wg2,  const float* __restrict__ bg2,
                     const float* __restrict__ qws,  const float* __restrict__ kws,
                     const float* __restrict__ vws,  const float* __restrict__ scws,
                     float* __restrict__ out)
{
  __shared__ float xyzs[512 * 3];
  __shared__ float wp1s[3 * 33];
  __shared__ float bp1s[33];
  __shared__ float bp2s[64], bg1s[64], bg2s[64], qrow[64];
  __shared__ float xi[3], sc_s[3], sh_s[3];
  __shared__ _Float16 Wp2t[64 * STRH];   // transposed, K padded 33->64 with zeros
  __shared__ _Float16 Wg1t[64 * STRH];
  __shared__ _Float16 Wg2t[64 * STRH];
  __shared__ _Float16 bufA[JCH * STRH];  // pe-hidden, later h
  __shared__ _Float16 bufB[JCH * STRH];  // attn_in
  __shared__ float    bufG[JCH * STRG];  // k-chunk, then g (f32 for softmax)
  __shared__ float    red_m[4 * 64], red_s[4 * 64], red_a[4 * 64];

  const int tid  = threadIdx.x;
  const int lane = tid & 31;
  const int wave = tid >> 5;
  const int bi = blockIdx.x;
  const int b  = bi >> 9;
  const int i  = bi & 511;

  const float* kp = kws + (size_t)b * 512 * 64;
  const float* vp = vws + (size_t)b * 512 * 64;

  // ---- stage parameters into LDS ----
  for (int idx = tid; idx < 512 * 3; idx += NTHREADS)
    xyzs[idx] = xyz[(size_t)b * 512 * 3 + idx];
  if (tid < 3) {
    xi[tid]   = xyz[((size_t)b * 512 + i) * 3 + tid];
    sc_s[tid] = scws[tid];
    sh_s[tid] = scws[4 + tid];
  }
  for (int idx = tid; idx < 99; idx += NTHREADS) wp1s[idx] = W_p1[idx];
  if (tid < 33) bp1s[tid] = b_p1[tid];
  if (tid < 64) {
    bp2s[tid] = b_p2[tid];
    bg1s[tid] = bg1[tid];
    bg2s[tid] = bg2[tid];
    qrow[tid] = qws[((size_t)b * 512 + i) * 64 + tid];
  }
  for (int idx = tid; idx < 64 * 64; idx += NTHREADS) {
    int n = idx >> 6, k = idx & 63;
    Wp2t[n * STRH + k] = (_Float16)((k < 33) ? W_p2[k * 64 + n] : 0.f);
    Wg1t[n * STRH + k] = (_Float16)Wg1[k * 64 + n];
    Wg2t[n * STRH + k] = (_Float16)Wg2[k * 64 + n];
  }
  __syncthreads();

  // per-thread online-softmax slice state: channel o = tid&63, slice s = tid>>6
  const int o_ch = tid & 63;
  const int s_sl = tid >> 6;
  float m_run = -3.0e38f, s_run = 0.f, a_run = 0.f;

  for (int ch = 0; ch < 512 / JCH; ++ch) {
    const int j0 = ch * JCH;

    // phase 1a: pe hidden (3 -> 33, relu), zero-padded to K=64, f16
    for (int idx = tid; idx < JCH * 64; idx += NTHREADS) {
      int j = idx >> 6, p = idx & 63;
      _Float16 hv = (_Float16)0.f;
      if (p < 33) {
        const float* xj = &xyzs[(j0 + j) * 3];
        float acc = bp1s[p];
#pragma unroll
        for (int c = 0; c < 3; ++c) {
          float d = (xi[c] - xj[c]) * sc_s[c] + sh_s[c];
          acc = fmaf(d, wp1s[c * 33 + p], acc);
        }
        hv = (_Float16)fmaxf(acc, 0.f);
      }
      bufA[j * STRH + p] = hv;
    }
    // phase 1b: stage k-chunk into bufG with coalesced global loads
    for (int idx = tid; idx < JCH * 64; idx += NTHREADS) {
      int j = idx >> 6, c = idx & 63;
      bufG[j * STRG + c] = kp[(size_t)(j0 + j) * 64 + c];
    }
    __syncthreads();

    // phase 2 (WMMA): pe = hp @ Wp2; attn = pe + b_p2 + q_i - k_j  -> f16 bufB
    {
      const int mt = wave;                 // one M-tile per wave, A loaded once
      v16h a0 = load_fragA(bufA, mt * 16, 0,  lane);
      v16h a1 = load_fragA(bufA, mt * 16, 32, lane);
      const int col0 = lane & 15, rb = (lane >> 4) << 3;
#pragma unroll
      for (int nt = 0; nt < 4; ++nt) {
        v16h w0 = load_fragB(Wp2t, nt * 16, 0,  lane);
        v16h w1 = load_fragB(Wp2t, nt * 16, 32, lane);
        v8f c = {};
        c = __builtin_amdgcn_wmma_f32_16x16x32_f16(false, a0, false, w0, (short)0, c, false, false);
        c = __builtin_amdgcn_wmma_f32_16x16x32_f16(false, a1, false, w1, (short)0, c, false, false);
        int col = nt * 16 + col0;
#pragma unroll
        for (int r = 0; r < 8; ++r) {
          int j = mt * 16 + rb + r;
          float val = c[r] + bp2s[col] + qrow[col] - bufG[j * STRG + col];
          bufB[j * STRH + col] = (_Float16)val;
        }
      }
    }
    __syncthreads();

    // phase 3 (WMMA): h = relu(attn @ Wg1 + bg1) -> f16 bufA (hp is dead)
    {
      const int mt = wave;
      v16h a0 = load_fragA(bufB, mt * 16, 0,  lane);
      v16h a1 = load_fragA(bufB, mt * 16, 32, lane);
      const int col0 = lane & 15, rb = (lane >> 4) << 3;
#pragma unroll
      for (int nt = 0; nt < 4; ++nt) {
        v16h w0 = load_fragB(Wg1t, nt * 16, 0,  lane);
        v16h w1 = load_fragB(Wg1t, nt * 16, 32, lane);
        v8f c = {};
        c = __builtin_amdgcn_wmma_f32_16x16x32_f16(false, a0, false, w0, (short)0, c, false, false);
        c = __builtin_amdgcn_wmma_f32_16x16x32_f16(false, a1, false, w1, (short)0, c, false, false);
        int col = nt * 16 + col0;
#pragma unroll
        for (int r = 0; r < 8; ++r) {
          int j = mt * 16 + rb + r;
          bufA[j * STRH + col] = (_Float16)fmaxf(c[r] + bg1s[col], 0.f);
        }
      }
    }
    __syncthreads();

    // phase 4 (WMMA): g = h @ Wg2 + bg2 -> f32 bufG (k-chunk is dead)
    {
      const int mt = wave;
      v16h a0 = load_fragA(bufA, mt * 16, 0,  lane);
      v16h a1 = load_fragA(bufA, mt * 16, 32, lane);
      const int col0 = lane & 15, rb = (lane >> 4) << 3;
#pragma unroll
      for (int nt = 0; nt < 4; ++nt) {
        v16h w0 = load_fragB(Wg2t, nt * 16, 0,  lane);
        v16h w1 = load_fragB(Wg2t, nt * 16, 32, lane);
        v8f c = {};
        c = __builtin_amdgcn_wmma_f32_16x16x32_f16(false, a0, false, w0, (short)0, c, false, false);
        c = __builtin_amdgcn_wmma_f32_16x16x32_f16(false, a1, false, w1, (short)0, c, false, false);
        int col = nt * 16 + col0;
#pragma unroll
        for (int r = 0; r < 8; ++r) {
          int j = mt * 16 + rb + r;
          bufG[j * STRG + col] = c[r] + bg2s[col];
        }
      }
    }
    __syncthreads();

    // phase 5: online softmax over this chunk's j-slice (j % 4 == s_sl),
    // all 8 waves active, fused with sum_j v[j,o]*exp(g)
    {
      float mloc = -3.0e38f;
#pragma unroll 4
      for (int jj = 0; jj < JCH / 4; ++jj)
        mloc = fmaxf(mloc, bufG[(4 * jj + s_sl) * STRG + o_ch]);
      float mnew = fmaxf(m_run, mloc);
      float corr = __expf(m_run - mnew);
      s_run *= corr; a_run *= corr;
#pragma unroll 4
      for (int jj = 0; jj < JCH / 4; ++jj) {
        float e = __expf(bufG[(4 * jj + s_sl) * STRG + o_ch] - mnew);
        s_run += e;
        a_run = fmaf(e, vp[(size_t)(j0 + 4 * jj + s_sl) * 64 + o_ch], a_run);
      }
      m_run = mnew;
    }
    __syncthreads();
  }

  // merge the 4 slice-partials per channel and emit y = acc / sum
  red_m[s_sl * 64 + o_ch] = m_run;
  red_s[s_sl * 64 + o_ch] = s_run;
  red_a[s_sl * 64 + o_ch] = a_run;
  __syncthreads();
  if (tid < 64) {
    float m = red_m[tid];
#pragma unroll
    for (int s = 1; s < 4; ++s) m = fmaxf(m, red_m[s * 64 + tid]);
    float S = 0.f, A = 0.f;
#pragma unroll
    for (int s = 0; s < 4; ++s) {
      float cc = __expf(red_m[s * 64 + tid] - m);
      S = fmaf(red_s[s * 64 + tid], cc, S);
      A = fmaf(red_a[s * 64 + tid], cc, A);
    }
    out[(size_t)bi * 64 + tid] = A / S;
  }
}

extern "C" void kernel_launch(void* const* d_in, const int* in_sizes, int n_in,
                              void* d_out, int out_size, void* d_ws, size_t ws_size,
                              hipStream_t stream) {
  (void)in_sizes; (void)n_in; (void)out_size; (void)ws_size;
  const float* xyz    = (const float*)d_in[0];
  const float* points = (const float*)d_in[1];
  const float* bn_g   = (const float*)d_in[2];
  const float* bn_b   = (const float*)d_in[3];
  const float* W_p1   = (const float*)d_in[4];
  const float* b_p1   = (const float*)d_in[5];
  const float* W_p2   = (const float*)d_in[6];
  const float* b_p2   = (const float*)d_in[7];
  const float* Wq     = (const float*)d_in[8];
  const float* bq     = (const float*)d_in[9];
  const float* Wk     = (const float*)d_in[10];
  const float* bk     = (const float*)d_in[11];
  const float* Wv     = (const float*)d_in[12];
  const float* bv     = (const float*)d_in[13];
  const float* Wg1    = (const float*)d_in[14];
  const float* bg1    = (const float*)d_in[15];
  const float* Wg2    = (const float*)d_in[16];
  const float* bg2    = (const float*)d_in[17];

  float* ws   = (float*)d_ws;
  float* scws = ws;                    // 16 floats: scale[3], pad, shift[3]
  float* qws  = ws + 16;
  float* kws  = qws + 2048 * 64;
  float* vws  = kws + 2048 * 64;
  float* out  = (float*)d_out;

  bn_stats_kernel<<<1, 32, 0, stream>>>(xyz, bn_g, bn_b, scws);
  qkv_kernel<<<2048, 192, 0, stream>>>(points, Wq, bq, Wk, bk, Wv, bv, qws, kws, vws);
  pt_layer_kernel<<<2048, NTHREADS, 0, stream>>>(xyz, W_p1, b_p1, W_p2, b_p2,
                                                 Wg1, bg1, Wg2, bg2,
                                                 qws, kws, vws, scws, out);
}
